// MultiHeadAttention_29119878267642
// MI455X (gfx1250) — compile-verified, matrix operand order checked by reading, not run on hardware
//
#include <hip/hip_runtime.h>
#include <hip/hip_bf16.h>

// Problem constants
#define BB 4
#define SS 2048
#define EE 1024
#define HH 16
#define DD 64
#define MM (BB * SS)     // 8192 rows (b*s)
#define NQKV (3 * EE)    // 3072
#define KE EE            // 1024 reduction dim

typedef __attribute__((ext_vector_type(16))) __bf16        v16bf;
typedef __attribute__((ext_vector_type(8)))  float         v8f;
typedef __attribute__((ext_vector_type(4)))  unsigned int  u32x4;
typedef __attribute__((ext_vector_type(4)))  int           v4i;

union FragU {
  v16bf          v;
  u32x4          q[2];
  unsigned short s[16];
};

__device__ __forceinline__ v8f zero8() {
  v8f z;
#pragma unroll
  for (int i = 0; i < 8; ++i) z[i] = 0.0f;
  return z;
}

// fp32 -> bf16 round-to-nearest-even
__device__ __forceinline__ unsigned short f2bf(float f) {
  union { float f; unsigned int u; } c;
  c.f = f;
  unsigned int u = c.u;
  unsigned int r = u + 0x7FFFu + ((u >> 16) & 1u);
  return (unsigned short)(r >> 16);
}

// Load a 16-bit A/B WMMA fragment (16x32 tile, one row/col per lane).
// Per CDNA5 ISA: lanes 0-15 hold K {0..7,16..23}, lanes 16-31 hold K {8..15,24..31}.
// p points at (row-start + k-base); koff = 0 (lane<16) or 8 (lane>=16).
// Works for global or LDS-backed pointers (compiler picks b128 / ds_load_b128).
__device__ __forceinline__ v16bf load_frag(const unsigned short* p, int koff) {
  FragU f;
  f.q[0] = *reinterpret_cast<const u32x4*>(p + koff);        // K koff..koff+7
  f.q[1] = *reinterpret_cast<const u32x4*>(p + koff + 16);   // K 16+koff..16+koff+7
  return f.v;
}

__device__ __forceinline__ v8f wmma_bf16(v16bf a, v16bf b, v8f c) {
  // (neg_a, A, neg_b, B, c_mod, C, reuse_a, reuse_b)
  return __builtin_amdgcn_wmma_f32_16x16x32_bf16(false, a, false, b, (short)0, c,
                                                 false, false);
}

// ---- global -> LDS staging (async-to-LDS when the toolchain exposes it) ----
#if defined(__AMDGCN__)
#  if __has_builtin(__builtin_amdgcn_global_load_async_to_lds_b128) && \
      __has_builtin(__builtin_amdgcn_s_wait_asynccnt)
#    define USE_ASYNC_LDS 1
#  endif
#endif

#ifdef USE_ASYNC_LDS
typedef __attribute__((address_space(1))) v4i gv4i;  // global int4
typedef __attribute__((address_space(3))) v4i lv4i;  // LDS int4
#endif

__device__ __forceinline__ void stage_b128(const unsigned short* src,
                                           unsigned short* dst /* LDS */) {
#ifdef USE_ASYNC_LDS
  // generic->as(1) is identity; generic->as(3): ISA maps LDS addr = addr[31:0]
  __builtin_amdgcn_global_load_async_to_lds_b128(
      (gv4i*)(unsigned long long)src,
      (lv4i*)(unsigned int)(unsigned long long)dst, 0, 0);
  __builtin_amdgcn_s_wait_asynccnt(0);
#else
  *reinterpret_cast<u32x4*>(dst) = *reinterpret_cast<const u32x4*>(src);
#endif
}

// ---------------------------------------------------------------- converts
__global__ void k_cvt(const float* __restrict__ src, unsigned short* __restrict__ dst,
                      int n) {
  for (int i = blockIdx.x * blockDim.x + threadIdx.x; i < n;
       i += gridDim.x * blockDim.x)
    dst[i] = f2bf(src[i]);
}

// src is [K, N] row-major fp32; dst is [N, K] bf16 (transposed, so B-fragments
// become contiguous b128 loads).
__global__ void k_cvt_T(const float* __restrict__ src, unsigned short* __restrict__ dst,
                        int K, int N) {
  int tot = K * N;
  for (int i = blockIdx.x * blockDim.x + threadIdx.x; i < tot;
       i += gridDim.x * blockDim.x) {
    int k = i / N;
    int n = i - k * N;
    dst[(size_t)n * K + k] = f2bf(src[i]);
  }
}

// LDS tile for GEMMs: 64 N-rows x 32 k, padded to 40 elems/row (80B) so the
// 64B-stride fragment reads destagger across banks. 5 KB.
#define BROW 40

// ---------------------------------------------------------- QKV projection
// Block = 256 thr (8 waves) computes 128 rows x 64 cols. All 8 waves share one
// LDS-staged B tile per k-step; each wave owns a 16-row A slice.
// Result scattered into Q[b,h,s,d], K[b,h,s,d], Vt[b,h,d,s] as bf16 (+bias).
__global__ void k_gemm_qkv(const unsigned short* __restrict__ xb,
                           const unsigned short* __restrict__ wT,
                           const float* __restrict__ bias,
                           unsigned short* __restrict__ Qb,
                           unsigned short* __restrict__ Kb,
                           unsigned short* __restrict__ Vt) {
  __shared__ unsigned short bsh[64 * BROW];
  const int NTB = NQKV / 64;  // 48 column blocks
  int tid = threadIdx.x;
  int w = tid >> 5, lane = tid & 31;
  int mblk = blockIdx.x / NTB;
  int nblk = blockIdx.x - mblk * NTB;
  int mb = mblk * 128 + w * 16;
  int nb = nblk * 64;
  int half = lane >> 4, lr = lane & 15;
  int koff = half ? 8 : 0;

  // staging assignment: one b128 per thread per k-step
  int n_local = tid >> 2;          // 0..63
  int kchunk = (tid & 3) * 8;      // 0,8,16,24
  const unsigned short* bsrc = wT + (size_t)(nb + n_local) * KE + kchunk;
  unsigned short* bdst = bsh + n_local * BROW + kchunk;

  v8f acc[4];
#pragma unroll
  for (int t = 0; t < 4; ++t) acc[t] = zero8();

  const unsigned short* arow = xb + (size_t)(mb + lr) * KE;
  for (int k = 0; k < KE; k += 32) {
    stage_b128(bsrc + k, bdst);
    __syncthreads();
    v16bf a = load_frag(arow + k, koff);
#pragma unroll
    for (int t = 0; t < 4; ++t) {
      v16bf b = load_frag(bsh + (16 * t + lr) * BROW, koff);
      acc[t] = wmma_bf16(a, b, acc[t]);
    }
    __syncthreads();
  }

#pragma unroll
  for (int t = 0; t < 4; ++t) {
    int n = nb + 16 * t + lr;   // fixed per lane per tile
    float bv = bias[n];
    int sec = n >> 10;          // 0=q 1=k 2=v
    int nn = n & 1023;
    int h = nn >> 6, d = nn & 63;
#pragma unroll
    for (int r = 0; r < 8; ++r) {
      int m = mb + r + 8 * half;      // C layout: VGPR r, lane half
      int b = m >> 11, s = m & 2047;  // S = 2048
      unsigned short val = f2bf(acc[t][r] + bv);
      size_t bh = (size_t)(b * HH + h);
      if (sec == 0)      Qb[(bh * SS + s) * DD + d] = val;
      else if (sec == 1) Kb[(bh * SS + s) * DD + d] = val;
      else               Vt[(bh * DD + d) * SS + s] = val;  // V transposed
    }
  }
}

// ------------------------------------------------------------- attention
// One wave per 16-query tile. Transposed-score formulation:
//   S^T = K_tile * Q^T   (C layout: lane = query column, regs = key rows)
//   O^T = V^T * P^T      (P^T packs from score C-regs with NO lane movement)
__global__ void k_attn(const unsigned short* __restrict__ Qb,
                       const unsigned short* __restrict__ Kb,
                       const unsigned short* __restrict__ Vt,
                       unsigned short* __restrict__ Ob) {
  int wid  = (blockIdx.x * blockDim.x + threadIdx.x) >> 5;
  int lane = threadIdx.x & 31;
  int qt = wid & (SS / 16 - 1);  // 0..127
  int bh = wid >> 7;             // 0..63
  int qb = qt * 16;
  int half = lane >> 4, lr = lane & 15;
  int koff = half ? 8 : 0;

  const unsigned short* Qh = Qb + (size_t)bh * SS * DD;
  const unsigned short* Kh = Kb + (size_t)bh * SS * DD;
  const unsigned short* Vh = Vt + (size_t)bh * DD * SS;

  // Q^T B-fragments for this wave's 16 queries (reused every key block)
  const unsigned short* qrow = Qh + (size_t)(qb + lr) * DD;
  v16bf bq0 = load_frag(qrow, koff);       // d 0..31
  v16bf bq1 = load_frag(qrow + 32, koff);  // d 32..63
  int q_idx = qb + lr;                      // this lane's query

  float m_run = -3.0e38f, l_run = 0.0f;
  v8f o[4];
#pragma unroll
  for (int t = 0; t < 4; ++t) o[t] = zero8();

  for (int jb = 0; jb < qb + 16; jb += 32) {  // causal: keys 0..qb+15
    // scores^T for 32 keys = two 16x16 C tiles, each K-dim 64 (2 chained wmma)
    const unsigned short* k0 = Kh + (size_t)(jb + lr) * DD;
    const unsigned short* k1 = k0 + 16 * DD;
    __builtin_prefetch(k0 + 32 * DD, 0, 0);  // next key block (speculative)
    v8f s0 = zero8(), s1 = zero8();
    s0 = wmma_bf16(load_frag(k0, koff),      bq0, s0);
    s0 = wmma_bf16(load_frag(k0 + 32, koff), bq1, s0);
    s1 = wmma_bf16(load_frag(k1, koff),      bq0, s1);
    s1 = wmma_bf16(load_frag(k1 + 32, koff), bq1, s1);

    // scale + causal mask; per-lane block max over its 16 key rows
    float bm = -3.0e38f;
#pragma unroll
    for (int r = 0; r < 8; ++r) {
      int key0 = jb + r + 8 * half;
      s0[r] = (key0 <= q_idx)      ? s0[r] * 0.125f : -3.0e38f;  // 1/sqrt(64)
      s1[r] = (key0 + 16 <= q_idx) ? s1[r] * 0.125f : -3.0e38f;
      bm = fmaxf(bm, fmaxf(s0[r], s1[r]));
    }
    bm = fmaxf(bm, __shfl_xor(bm, 16, 32));  // other half holds the other 16 keys
    float m_new = fmaxf(m_run, bm);
    float scale = __expf(m_run - m_new);

    // P = exp(S - m); pack P^T B-fragment directly from C regs (same-lane!)
    FragU bp;
    float rs = 0.0f;
#pragma unroll
    for (int r = 0; r < 8; ++r) {
      float p0 = __expf(s0[r] - m_new);
      float p1 = __expf(s1[r] - m_new);
      rs += p0 + p1;
      bp.s[r]     = f2bf(p0);
      bp.s[r + 8] = f2bf(p1);
    }
    rs += __shfl_xor(rs, 16, 32);
    l_run = l_run * scale + rs;
    m_run = m_new;

    // O^T += V^T * P^T  (4 d-tiles of 16; V^T A-frags are contiguous b128s)
#pragma unroll
    for (int t = 0; t < 4; ++t) {
      o[t] = o[t] * scale;
      const unsigned short* vrow = Vh + (size_t)(16 * t + lr) * SS + jb;
      o[t] = wmma_bf16(load_frag(vrow, koff), bp.v, o[t]);
    }
  }

  float inv = 1.0f / l_run;
  int b = bh >> 4, h = bh & 15;
#pragma unroll
  for (int t = 0; t < 4; ++t)
#pragma unroll
    for (int r = 0; r < 8; ++r) {
      int d = 16 * t + r + 8 * half;
      Ob[((size_t)(b * SS + q_idx)) * EE + h * DD + d] = f2bf(o[t][r] * inv);
    }
}

// --------------------------------------------------------- output proj
__global__ void k_gemm_proj(const unsigned short* __restrict__ ab,
                            const unsigned short* __restrict__ wT,
                            const float* __restrict__ bias,
                            float* __restrict__ out) {
  __shared__ unsigned short bsh[64 * BROW];
  const int NTB = EE / 64;  // 16 column blocks
  int tid = threadIdx.x;
  int w = tid >> 5, lane = tid & 31;
  int mblk = blockIdx.x / NTB;
  int nblk = blockIdx.x - mblk * NTB;
  int mb = mblk * 128 + w * 16;
  int nb = nblk * 64;
  int half = lane >> 4, lr = lane & 15;
  int koff = half ? 8 : 0;

  int n_local = tid >> 2;
  int kchunk = (tid & 3) * 8;
  const unsigned short* bsrc = wT + (size_t)(nb + n_local) * KE + kchunk;
  unsigned short* bdst = bsh + n_local * BROW + kchunk;

  v8f acc[4];
#pragma unroll
  for (int t = 0; t < 4; ++t) acc[t] = zero8();

  const unsigned short* arow = ab + (size_t)(mb + lr) * KE;
  for (int k = 0; k < KE; k += 32) {
    stage_b128(bsrc + k, bdst);
    __syncthreads();
    v16bf a = load_frag(arow + k, koff);
#pragma unroll
    for (int t = 0; t < 4; ++t) {
      v16bf b = load_frag(bsh + (16 * t + lr) * BROW, koff);
      acc[t] = wmma_bf16(a, b, acc[t]);
    }
    __syncthreads();
  }

#pragma unroll
  for (int t = 0; t < 4; ++t) {
    int n = nb + 16 * t + lr;
    float bv = bias[n];
#pragma unroll
    for (int r = 0; r < 8; ++r) {
      int m = mb + r + 8 * half;
      out[(size_t)m * EE + n] = acc[t][r] + bv;
    }
  }
}

// ------------------------------------------------------------------ host
extern "C" void kernel_launch(void* const* d_in, const int* in_sizes, int n_in,
                              void* d_out, int out_size, void* d_ws, size_t ws_size,
                              hipStream_t stream) {
  const float* x      = (const float*)d_in[0];
  const float* W_attn = (const float*)d_in[1];
  const float* b_attn = (const float*)d_in[2];
  const float* W_proj = (const float*)d_in[3];
  const float* b_proj = (const float*)d_in[4];
  float* out = (float*)d_out;

  char* ws = (char*)d_ws;
  size_t off = 0;
  auto take = [&](size_t elems) -> unsigned short* {
    unsigned short* p = (unsigned short*)(ws + off);
    off += ((elems * sizeof(unsigned short)) + 255) & ~(size_t)255;
    return p;
  };
  unsigned short* xb     = take((size_t)MM * KE);          // x as bf16
  unsigned short* wqkvT  = take((size_t)NQKV * KE);        // W_attn^T bf16
  unsigned short* wprojT = take((size_t)EE * KE);          // W_proj^T bf16
  unsigned short* Qb     = take((size_t)BB * HH * SS * DD);
  unsigned short* Kb     = take((size_t)BB * HH * SS * DD);
  unsigned short* Vt     = take((size_t)BB * HH * DD * SS);  // V transposed
  unsigned short* Ob     = take((size_t)MM * EE);            // attention out bf16

  k_cvt<<<2048, 256, 0, stream>>>(x, xb, MM * KE);
  k_cvt_T<<<2048, 256, 0, stream>>>(W_attn, wqkvT, KE, NQKV);
  k_cvt_T<<<1024, 256, 0, stream>>>(W_proj, wprojT, KE, EE);

  k_gemm_qkv<<<(MM / 128) * (NQKV / 64), 256, 0, stream>>>(xb, wqkvT, b_attn,
                                                           Qb, Kb, Vt);
  k_attn<<<BB * HH * (SS / 16) / 8, 256, 0, stream>>>(Qb, Kb, Vt, Ob);
  k_gemm_proj<<<(MM / 128) * (EE / 64), 256, 0, stream>>>(Ob, wprojT, b_proj,
                                                          out);
}
